// SUBNET_LPV_58179626992073
// MI455X (gfx1250) — matrix operand order repaired and checked by reading, not compiled, not run on hardware
//
#include <hip/hip_runtime.h>

typedef _Float16 h16;
typedef __attribute__((ext_vector_type(16))) _Float16 v16h;
typedef __attribute__((ext_vector_type(8)))  _Float16 v8h;
typedef __attribute__((ext_vector_type(8)))  float    v8f;

#define WAVES  4
#define BLOCK  (WAVES * 32)
#define NBAT   2048
#define TSTEPS 512
#define EIN    120
#define EKP    128     // encoder input K padded
#define KBIG   288     // 9 slots (8 p + 1 lin) * 32 padded inputs

// ---- fragment loaders (CDNA5 ISA 7.12.2 16-bit layouts) -------------------
// A-matrix 16x32: lanes0-15 row=M K{0..7,16..23}, lanes16-31 row=M K{8..15,24..31}
__device__ __forceinline__ v16h ld_fragA(const h16* base, int S, int row, int hi, int kbase) {
    const h16* p = base + row * S + kbase + hi * 8;
    v8h a = *(const v8h*)(p);
    v8h b = *(const v8h*)(p + 16);
    v16h r;
#pragma unroll
    for (int e = 0; e < 8; ++e) { r[e] = a[e]; r[e + 8] = b[e]; }
    return r;
}
// B-matrix 32x16 (stored transposed [N][K] in LDS): lane = column N,
// lanes0-15 hold K{0..7,16..23}, lanes16-31 hold K{8..15,24..31} (matches A)
__device__ __forceinline__ v16h ld_fragB(const h16* base, int S, int col, int hi, int kbase) {
    const h16* p = base + col * S + kbase + hi * 8;
    v8h a = *(const v8h*)(p);
    v8h b = *(const v8h*)(p + 16);
    v16h r;
#pragma unroll
    for (int e = 0; e < 8; ++e) { r[e] = a[e]; r[e + 8] = b[e]; }
    return r;
}
__device__ __forceinline__ v8f wmma32(v16h a, v16h b, v8f c) {
    return __builtin_amdgcn_wmma_f32_16x16x32_f16(false, a, false, b, (short)0, c, false, false);
}
__device__ __forceinline__ v8f splat8(float x) {
    v8f r;
#pragma unroll
    for (int e = 0; e < 8; ++e) r[e] = x;
    return r;
}

__global__ __launch_bounds__(BLOCK) void subnet_lpv_kernel(
    const float* __restrict__ upast, const float* __restrict__ ypast, const float* __restrict__ ufuture,
    const float* __restrict__ eW0, const float* __restrict__ eb0i, const float* __restrict__ eW1,
    const float* __restrict__ eb1i, const float* __restrict__ eW2, const float* __restrict__ eb2i,
    const float* __restrict__ eWr,
    const float* __restrict__ sW0, const float* __restrict__ sb0i, const float* __restrict__ sW1,
    const float* __restrict__ sb1i, const float* __restrict__ sW2, const float* __restrict__ sb2i,
    const float* __restrict__ sWr,
    const float* __restrict__ A_lin, const float* __restrict__ A_p,
    const float* __restrict__ B_lin, const float* __restrict__ B_p,
    const float* __restrict__ C_lin, const float* __restrict__ C_p,
    const float* __restrict__ D_lin, const float* __restrict__ D_p,
    float* __restrict__ yout)
{
    // ---------------- shared weights (operand-ready, transposed, f16) ------
    __shared__ __align__(16) h16 sW0T[64][32];
    __shared__ __align__(16) h16 sW1T[64][64];
    __shared__ __align__(16) h16 sW2T[16][64];
    __shared__ __align__(16) h16 sWrT[16][32];
    __shared__ __align__(16) h16 gW0T[64][EKP];
    __shared__ __align__(16) h16 gW1T[64][64];
    __shared__ __align__(16) h16 gW2T[16][64];
    __shared__ __align__(16) h16 gWrT[16][EKP];
    __shared__ __align__(16) h16 sWbT[32][KBIG];   // merged bilinear [o][s*32+i]
    __shared__ float lb0[64], lb1[64], lb2[16];
    __shared__ float kb0[64], kb1[64], kb2[16];
    // ---------------- per-wave working buffers -----------------------------
    __shared__ __align__(16) h16 sV [WAVES][16][32];   // [x(16) u(3) 0-pad]
    __shared__ __align__(16) h16 sH1[WAVES][16][64];
    __shared__ __align__(16) h16 sH2[WAVES][16][64];
    __shared__ __align__(16) h16 sE [WAVES][16][EKP];
    __shared__            h16 sP [WAVES][16][8];

    const int tid = threadIdx.x;

    // ======== one-time weight preprocessing (whole workgroup) ==============
    for (int idx = tid; idx < 64 * 32; idx += BLOCK) {            // sch W0 [19][64]
        int n = idx >> 5, k = idx & 31;
        sW0T[n][k] = (h16)(k < 19 ? sW0[k * 64 + n] : 0.0f);
    }
    for (int idx = tid; idx < 64 * 64; idx += BLOCK) {            // sch W1 [64][64]
        int n = idx >> 6, k = idx & 63;
        sW1T[n][k] = (h16)sW1[k * 64 + n];
    }
    for (int idx = tid; idx < 16 * 64; idx += BLOCK) {            // sch W2 [64][8]
        int n = idx >> 6, k = idx & 63;
        sW2T[n][k] = (h16)(n < 8 ? sW2[k * 8 + n] : 0.0f);
    }
    for (int idx = tid; idx < 16 * 32; idx += BLOCK) {            // sch Wres [19][8]
        int n = idx >> 5, k = idx & 31;
        sWrT[n][k] = (h16)((n < 8 && k < 19) ? sWr[k * 8 + n] : 0.0f);
    }
    for (int idx = tid; idx < 64 * EKP; idx += BLOCK) {           // enc W0 [120][64]
        int n = idx >> 7, k = idx & 127;
        gW0T[n][k] = (h16)(k < EIN ? eW0[k * 64 + n] : 0.0f);
    }
    for (int idx = tid; idx < 64 * 64; idx += BLOCK) {            // enc W1
        int n = idx >> 6, k = idx & 63;
        gW1T[n][k] = (h16)eW1[k * 64 + n];
    }
    for (int idx = tid; idx < 16 * 64; idx += BLOCK) {            // enc W2 [64][16]
        int n = idx >> 6, k = idx & 63;
        gW2T[n][k] = (h16)eW2[k * 16 + n];
    }
    for (int idx = tid; idx < 16 * EKP; idx += BLOCK) {           // enc Wres [120][16]
        int n = idx >> 7, k = idx & 127;
        gWrT[n][k] = (h16)(k < EIN ? eWr[k * 16 + n] : 0.0f);
    }
    for (int idx = tid; idx < 64; idx += BLOCK) { lb0[idx] = sb0i[idx]; lb1[idx] = sb1i[idx];
                                                  kb0[idx] = eb0i[idx]; kb1[idx] = eb1i[idx]; }
    for (int idx = tid; idx < 16; idx += BLOCK) { lb2[idx] = idx < 8 ? sb2i[idx] : 0.0f;
                                                  kb2[idx] = eb2i[idx]; }
    // merged bilinear weight: out[b,o] = sum_f Wb[f,o] * zhat[b,f], f = s*32+i
    for (int idx = tid; idx < 32 * KBIG; idx += BLOCK) {
        int o = idx / KBIG, f = idx - o * KBIG;
        int s = f >> 5, i = f & 31;
        float v = 0.0f;
        if (o < 16) {                                   // xn rows: A | Bm
            if (i < 16)      v = (s < 8) ? A_p[(s * 16 + o) * 16 + i] : A_lin[o * 16 + i];
            else if (i < 19) v = (s < 8) ? B_p[(s * 16 + o) * 3 + (i - 16)] : B_lin[o * 3 + (i - 16)];
        } else if (o < 19) {                            // y rows: C | D
            int oy = o - 16;
            if (i < 16)      v = (s < 8) ? C_p[(s * 3 + oy) * 16 + i] : C_lin[oy * 16 + i];
            else if (i < 19) v = (s < 8) ? D_p[(s * 3 + oy) * 3 + (i - 16)] : D_lin[oy * 3 + (i - 16)];
        }
        sWbT[o][f] = (h16)v;
    }
    __syncthreads();

    // ======== per-wave setup ==============================================
    const int wv   = tid >> 5;
    const int lane = tid & 31;
    const int ln   = lane & 15;
    const int hi   = lane >> 4;
    const int base = (blockIdx.x * WAVES + wv) * 16;   // 32 blocks * 4 waves * 16 rows = 2048

    // stage encoder input [16][EKP] (wave-private, no barrier needed)
    for (int idx = lane; idx < 16 * EIN; idx += 32) {
        int r = idx / EIN, k = idx - r * EIN;
        float v = (k < 60) ? upast[(base + r) * 60 + k] : ypast[(base + r) * 60 + (k - 60)];
        sE[wv][r][k] = (h16)v;
    }
    for (int idx = lane; idx < 16 * (EKP - EIN); idx += 32) {
        int r = idx >> 3, k = EIN + (idx & 7);
        sE[wv][r][k] = (h16)0.0f;
    }
    for (int idx = lane; idx < 16 * 16; idx += 32) {   // zero v pad cols 16..31
        int r = idx >> 4, c = 16 + (idx & 15);
        sV[wv][r][c] = (h16)0.0f;
    }

    // ======== encoder MLP -> x0 ===========================================
    {
        v16h ef[4];
#pragma unroll
        for (int kt = 0; kt < 4; ++kt) ef[kt] = ld_fragA(&sE[wv][0][0], EKP, ln, hi, kt * 32);
#pragma unroll
        for (int nt = 0; nt < 4; ++nt) {               // layer 1
            v8f acc = splat8(kb0[nt * 16 + ln]);
#pragma unroll
            for (int kt = 0; kt < 4; ++kt)
                acc = wmma32(ef[kt], ld_fragB(&gW0T[0][0], EKP, nt * 16 + ln, hi, kt * 32), acc);
#pragma unroll
            for (int j = 0; j < 8; ++j)
                sH1[wv][j + 8 * hi][nt * 16 + ln] = (h16)fmaxf(acc[j], 0.0f);
        }
        v16h h1a = ld_fragA(&sH1[wv][0][0], 64, ln, hi, 0);
        v16h h1b = ld_fragA(&sH1[wv][0][0], 64, ln, hi, 32);
#pragma unroll
        for (int nt = 0; nt < 4; ++nt) {               // layer 2
            v8f acc = splat8(kb1[nt * 16 + ln]);
            acc = wmma32(h1a, ld_fragB(&gW1T[0][0], 64, nt * 16 + ln, hi, 0), acc);
            acc = wmma32(h1b, ld_fragB(&gW1T[0][0], 64, nt * 16 + ln, hi, 32), acc);
#pragma unroll
            for (int j = 0; j < 8; ++j)
                sH2[wv][j + 8 * hi][nt * 16 + ln] = (h16)fmaxf(acc[j], 0.0f);
        }
        v16h h2a = ld_fragA(&sH2[wv][0][0], 64, ln, hi, 0);
        v16h h2b = ld_fragA(&sH2[wv][0][0], 64, ln, hi, 32);
        v8f acc = splat8(kb2[ln]);                     // layer 3 + residual
        acc = wmma32(h2a, ld_fragB(&gW2T[0][0], 64, ln, hi, 0), acc);
        acc = wmma32(h2b, ld_fragB(&gW2T[0][0], 64, ln, hi, 32), acc);
#pragma unroll
        for (int kt = 0; kt < 4; ++kt)
            acc = wmma32(ef[kt], ld_fragB(&gWrT[0][0], EKP, ln, hi, kt * 32), acc);
#pragma unroll
        for (int j = 0; j < 8; ++j)
            sV[wv][j + 8 * hi][ln] = (h16)acc[j];      // x0 -> state
    }

    // ======== T-step recurrence (no barriers: all LDS traffic wave-private)
    for (int t = 0; t < TSTEPS; ++t) {
        if (!hi) {                                     // lanes 0..15: load u_t for row ln
            const float* uu = ufuture + ((size_t)(base + ln) * TSTEPS + t) * 3;
            sV[wv][ln][16] = (h16)uu[0];
            sV[wv][ln][17] = (h16)uu[1];
            sV[wv][ln][18] = (h16)uu[2];
            if (t + 1 < TSTEPS) __builtin_prefetch(uu + 3, 0, 0);   // global_prefetch_b8
        }
        v16h vf = ld_fragA(&sV[wv][0][0], 32, ln, hi, 0);

        // scheduling MLP layer 1: [16,32] @ [32,64]
#pragma unroll
        for (int nt = 0; nt < 4; ++nt) {
            v8f acc = splat8(lb0[nt * 16 + ln]);
            acc = wmma32(vf, ld_fragB(&sW0T[0][0], 32, nt * 16 + ln, hi, 0), acc);
#pragma unroll
            for (int j = 0; j < 8; ++j)
                sH1[wv][j + 8 * hi][nt * 16 + ln] = (h16)fmaxf(acc[j], 0.0f);
        }
        v16h h1a = ld_fragA(&sH1[wv][0][0], 64, ln, hi, 0);
        v16h h1b = ld_fragA(&sH1[wv][0][0], 64, ln, hi, 32);
        // layer 2: [16,64] @ [64,64]
#pragma unroll
        for (int nt = 0; nt < 4; ++nt) {
            v8f acc = splat8(lb1[nt * 16 + ln]);
            acc = wmma32(h1a, ld_fragB(&sW1T[0][0], 64, nt * 16 + ln, hi, 0), acc);
            acc = wmma32(h1b, ld_fragB(&sW1T[0][0], 64, nt * 16 + ln, hi, 32), acc);
#pragma unroll
            for (int j = 0; j < 8; ++j)
                sH2[wv][j + 8 * hi][nt * 16 + ln] = (h16)fmaxf(acc[j], 0.0f);
        }
        v16h h2a = ld_fragA(&sH2[wv][0][0], 64, ln, hi, 0);
        v16h h2b = ld_fragA(&sH2[wv][0][0], 64, ln, hi, 32);
        // p = h2 @ W2 + v @ Wres + b2   (cols 8..15 padded to zero)
        v8f pac = splat8(lb2[ln]);
        pac = wmma32(h2a, ld_fragB(&sW2T[0][0], 64, ln, hi, 0), pac);
        pac = wmma32(h2b, ld_fragB(&sW2T[0][0], 64, ln, hi, 32), pac);
        pac = wmma32(vf,  ld_fragB(&sWrT[0][0], 32, ln, hi, 0), pac);
        if (ln < 8) {
#pragma unroll
            for (int j = 0; j < 8; ++j)
                sP[wv][j + 8 * hi][ln] = (h16)pac[j];
        }

        // bilinear state update as rank-expanded matmul:
        // z[b, s*32+i] = phat[b,s] * v[b,i];  [xn|y] = z @ Wbig
        v8f o0 = splat8(0.0f), o1 = splat8(0.0f);
#pragma unroll
        for (int s = 0; s < 9; ++s) {
            v16h zf;
            if (s < 8) {
                h16 ps = sP[wv][ln][s];
                v16h psp;
#pragma unroll
                for (int e = 0; e < 16; ++e) psp[e] = ps;
                zf = vf * psp;                         // v_pk_mul_f16 x8
            } else {
                zf = vf;                               // lin slot, phat = 1
            }
            o0 = wmma32(zf, ld_fragB(&sWbT[0][0], KBIG, ln,      hi, s * 32), o0);
            o1 = wmma32(zf, ld_fragB(&sWbT[0][0], KBIG, 16 + ln, hi, s * 32), o1);
        }
        // y (cols 16..18) -> global f32
        if (ln < 3) {
#pragma unroll
            for (int j = 0; j < 8; ++j)
                yout[((size_t)(base + j + 8 * hi) * TSTEPS + t) * 3 + ln] = o1[j];
        }
        // xn (cols 0..15) -> state for next step
#pragma unroll
        for (int j = 0; j < 8; ++j)
            sV[wv][j + 8 * hi][ln] = (h16)o0[j];
    }
}

extern "C" void kernel_launch(void* const* d_in, const int* in_sizes, int n_in,
                              void* d_out, int out_size, void* d_ws, size_t ws_size,
                              hipStream_t stream) {
    (void)in_sizes; (void)n_in; (void)d_ws; (void)ws_size; (void)out_size;
    const float* upast   = (const float*)d_in[0];
    const float* ypast   = (const float*)d_in[1];
    const float* ufuture = (const float*)d_in[2];
    const float* eW0 = (const float*)d_in[3];  const float* eb0 = (const float*)d_in[4];
    const float* eW1 = (const float*)d_in[5];  const float* eb1 = (const float*)d_in[6];
    const float* eW2 = (const float*)d_in[7];  const float* eb2 = (const float*)d_in[8];
    const float* eWr = (const float*)d_in[9];
    const float* sW0 = (const float*)d_in[10]; const float* sb0 = (const float*)d_in[11];
    const float* sW1 = (const float*)d_in[12]; const float* sb1 = (const float*)d_in[13];
    const float* sW2 = (const float*)d_in[14]; const float* sb2 = (const float*)d_in[15];
    const float* sWr = (const float*)d_in[16];
    const float* A_lin = (const float*)d_in[17]; const float* A_p = (const float*)d_in[18];
    const float* B_lin = (const float*)d_in[19]; const float* B_p = (const float*)d_in[20];
    const float* C_lin = (const float*)d_in[21]; const float* C_p = (const float*)d_in[22];
    const float* D_lin = (const float*)d_in[23]; const float* D_p = (const float*)d_in[24];
    float* yout = (float*)d_out;

    dim3 grid(NBAT / (WAVES * 16));   // 32 workgroups
    dim3 block(BLOCK);                // 128 threads = 4 waves (wave32)
    subnet_lpv_kernel<<<grid, block, 0, stream>>>(
        upast, ypast, ufuture,
        eW0, eb0, eW1, eb1, eW2, eb2, eWr,
        sW0, sb0, sW1, sb1, sW2, sb2, sWr,
        A_lin, A_p, B_lin, B_p, C_lin, C_p, D_lin, D_p,
        yout);
}
